// DGCNN_cls_img_28956669510144
// MI455X (gfx1250) — matrix operand order repaired and checked by reading, not compiled, not run on hardware
//
#include <hip/hip_runtime.h>
#include <hip/hip_bf16.h>

typedef __attribute__((ext_vector_type(16))) _Float16 v16h;
typedef __attribute__((ext_vector_type(8)))  float    v8f;

#define BDIM   16
#define NPTS   1024
#define KNN_K  20
#define NKCOLS (NPTS*KNN_K)
#define EPSV   1e-5f
#define NEG_INF (-3.0e38f)

__device__ __forceinline__ float lrelu(float x) { return x >= 0.f ? x : 0.2f * x; }

__device__ __forceinline__ void atomicMaxF(float* addr, float val) {
  unsigned int* ua = reinterpret_cast<unsigned int*>(addr);
  unsigned int old = *ua;
  while (__uint_as_float(old) < val) {
    unsigned int assumed = old;
    old = atomicCAS(ua, assumed, __float_as_uint(val));
    if (old == assumed) break;
  }
}

__device__ __forceinline__ v8f wmma_f16(v16h a, v16h b, v8f c) {
  return __builtin_amdgcn_wmma_f32_16x16x32_f16(false, a, false, b, (short)0, c,
                                                false, false);
}

// ---- weight packing: W (O x C2) fp32 -> f16 in WMMA A-fragment order --------
// Wp[((mtile*KCH + kc)*32 + lane)*16 + t], lane = half*16 + m,
// t -> K = half*16 + t  (contiguous order, matches B fragment reads)
__global__ void k_packA(const float* __restrict__ W, int C2, int KCH,
                        _Float16* __restrict__ Wp, int total) {
  int t = blockIdx.x * blockDim.x + threadIdx.x;
  if (t >= total) return;
  int kk = t & 15;
  int lane = (t >> 4) & 31;
  int kc = (t >> 9) % KCH;
  int mtile = (t >> 9) / KCH;
  int m = lane & 15, half = lane >> 4;
  int i = kk >> 1;
  int kb = half * 16 + kk; (void)i;
  int kcol = kc * 32 + kb;
  float v = (kcol < C2) ? W[(size_t)(mtile * 16 + m) * C2 + kcol] : 0.f;
  Wp[t] = (_Float16)v;
}

// ---- utility kernels --------------------------------------------------------
__global__ void k_fill(float* p, long n, float v) {
  long i = (long)blockIdx.x * blockDim.x + threadIdx.x;
  if (i < n) p[i] = v;
}

// x (B,5,N) -> xt (B,N,5)
__global__ void k_xt0(const float* __restrict__ x, float* __restrict__ xt) {
  int i = blockIdx.x * blockDim.x + threadIdx.x;
  if (i >= BDIM * 5 * NPTS) return;
  int n = i % NPTS; int c = (i / NPTS) % 5; int b = i / (5 * NPTS);
  xt[((size_t)(b * NPTS + n)) * 5 + c] = x[i];
}

// ---- KNN: per-thread point, top-20 of pd = -||xi-xj||^2 ---------------------
__global__ void __launch_bounds__(64) k_knn(const float* __restrict__ xt, int C,
                                            int* __restrict__ idxb) {
  __shared__ float own[64 * 128];
  __shared__ float chk[32 * 128];
  int tid = threadIdx.x;
  int b = blockIdx.x >> 4;                  // 16 blocks per batch
  int n = ((blockIdx.x & 15) << 6) + tid;
  const float* base = xt + (size_t)b * NPTS * C;
  for (int c = 0; c < C; ++c) own[tid * C + c] = base[(size_t)n * C + c];
  float bd[KNN_K]; int bi[KNN_K];
  for (int k = 0; k < KNN_K; ++k) { bd[k] = NEG_INF; bi[k] = 0; }
  for (int mc = 0; mc < NPTS / 32; ++mc) {
    __syncthreads();
    for (int e = tid; e < 32 * C; e += 64) chk[e] = base[(size_t)(mc * 32) * C + e];
    __syncthreads();
    if (mc + 1 < NPTS / 32)
      __builtin_prefetch(base + (size_t)((mc + 1) * 32) * C, 0, 3);
    for (int mm = 0; mm < 32; ++mm) {
      float s = 0.f;
      for (int c = 0; c < C; ++c) {
        float d = own[tid * C + c] - chk[mm * C + c];
        s += d * d;
      }
      float pd = -s;
      if (pd > bd[KNN_K - 1]) {
        int p = KNN_K - 1;
        while (p > 0 && bd[p - 1] < pd) {
          bd[p] = bd[p - 1]; bi[p] = bi[p - 1]; --p;
        }
        bd[p] = pd; bi[p] = mc * 32 + mm;
      }
    }
  }
  for (int k = 0; k < KNN_K; ++k)
    idxb[((size_t)(b * NPTS + n)) * KNN_K + k] = bi[k];
}

// ---- edge-conv GEMM: h = W(O x 2C) * feat(2C x (n,k)); fused max_k + stats --
// B panel stored fragment-ready: Bp[(kc*64 + col)*32 + k_in_chunk] so a lane's
// v16h is one contiguous 32B LDS read (-> ds_load_b128 x2).
template <int KCH>
__global__ void __launch_bounds__(128) k_ecgemm(
    const float* __restrict__ xt, int C, const _Float16* __restrict__ Wp, int O,
    const int* __restrict__ idxb, float* __restrict__ hmax,
    float* __restrict__ gsum, float* __restrict__ gsumsq) {
  __shared__ __align__(32) _Float16 Bp[KCH * 32 * 64];
  __shared__ float lsum[256];
  __shared__ float lsq[256];
  constexpr int CB = KCH * 32;
  int tid = threadIdx.x;
  int b = blockIdx.y;
  int colbase = blockIdx.x * 64;
  int C2 = 2 * C;
  const float* xtb = xt + (size_t)b * NPTS * C;
  const int* idb = idxb + (size_t)b * NPTS * KNN_K;
  for (int e = tid; e < 256; e += 128) { lsum[e] = 0.f; lsq[e] = 0.f; }
  // build the 2C x 64 edge-feature panel (f16, fragment-ready order)
  for (int e = tid; e < CB * 64; e += 128) {
    int c = e % CB;          // K index (padded C2)
    int j = e / CB;          // column within tile
    int colflat = colbase + j;
    float v = 0.f;
    if (c < C2 && colflat < NKCOLS) {
      int n = colflat / KNN_K, k = colflat % KNN_K;
      if (c < C) {
        int nb = idb[n * KNN_K + k];
        v = xtb[(size_t)nb * C + c] - xtb[(size_t)n * C + c];
      } else {
        v = xtb[(size_t)n * C + (c - C)];
      }
    }
    int kc = c >> 5, kin = c & 31;
    Bp[(kc * 64 + j) * 32 + kin] = (_Float16)v;
  }
  __syncthreads();
  int lane = tid & 31, wave = tid >> 5;
  int col0 = wave * 16;
  int n16 = lane & 15, khalf = lane >> 4;
  v16h bfr[KCH];
#pragma unroll
  for (int kc = 0; kc < KCH; ++kc)
    bfr[kc] = *(const v16h*)(Bp + (size_t)(kc * 64 + col0 + n16) * 32 + khalf * 16);
  int j = col0 + n16;
  int colflat = colbase + j;
  bool valid = colflat < NKCOLS;
  int n = colflat / KNN_K;
  int mt = O >> 4;
  for (int mtile = 0; mtile < mt; ++mtile) {
    v8f acc = {};
#pragma unroll
    for (int kc = 0; kc < KCH; ++kc) {
      v16h a = *(const v16h*)(Wp + (((size_t)(mtile * KCH + kc) * 32 + lane) << 4));
      acc = wmma_f16(a, bfr[kc], acc);
    }
    if (valid) {
#pragma unroll
      for (int r = 0; r < 8; ++r) {
        float v = acc[r];
        int o = mtile * 16 + r + (khalf << 3);
        atomicAdd(&lsum[o], v);
        atomicAdd(&lsq[o], v * v);
        atomicMaxF(&hmax[((size_t)(b * O + o)) * NPTS + n], v);
      }
    }
  }
  __syncthreads();
  if (tid < O) {
    atomicAdd(&gsum[tid], lsum[tid]);
    atomicAdd(&gsumsq[tid], lsq[tid]);
  }
}

__global__ void k_bnstats(const float* __restrict__ gsum, const float* __restrict__ gsq,
                          float count, int O, float* __restrict__ mean,
                          float* __restrict__ scale) {
  int o = blockIdx.x * blockDim.x + threadIdx.x;
  if (o >= O) return;
  float m = gsum[o] / count;
  float v = gsq[o] / count - m * m;
  mean[o] = m;
  scale[o] = rsqrtf(v + EPSV);
}

// apply bn+lrelu to per-(b,o,n) max; write f16 concat slice + next (b,n,c)
__global__ void k_ecapply(const float* __restrict__ hmax, int O,
                          const float* __restrict__ mean, const float* __restrict__ scale,
                          _Float16* __restrict__ catH, int coff,
                          float* __restrict__ xtn, int writeXt) {
  int i = blockIdx.x * blockDim.x + threadIdx.x;
  if (i >= BDIM * O * NPTS) return;
  int n = i % NPTS; int o = (i / NPTS) % O; int b = i / (O * NPTS);
  float y = lrelu((hmax[i] - mean[o]) * scale[o]);
  catH[((size_t)(b * 512 + coff + o)) * NPTS + n] = (_Float16)y;
  if (writeXt) xtn[((size_t)(b * NPTS + n)) * O + o] = y;
}

// ---- W5 GEMM (1024x512 @ 512xN per batch), 2 passes: stats, then pool -------
__global__ void __launch_bounds__(128) k_w5(
    const _Float16* __restrict__ catH, const _Float16* __restrict__ Wp, int pass,
    const float* __restrict__ mean, const float* __restrict__ scale,
    float* __restrict__ gsum, float* __restrict__ gsumsq, float* __restrict__ g) {
  __shared__ __align__(32) _Float16 Bp[512 * 16];
  int tid = threadIdx.x;
  int b = blockIdx.y;
  int lane = tid & 31, wave = tid >> 5;
  int n16 = lane & 15, khalf = lane >> 4;
  int mtile = blockIdx.x * 4 + wave;
  int row0 = mtile * 16;
  __builtin_amdgcn_s_wait_tensorcnt(0);
  __builtin_amdgcn_s_cluster_barrier();   // NOP outside a cluster
  v16h afr[16];
#pragma unroll
  for (int kc = 0; kc < 16; ++kc)
    afr[kc] = *(const v16h*)(Wp + (((size_t)(mtile * 16 + kc) * 32 + lane) << 4));
  float rmax[8], rsum[8], s1[8], s2[8];
#pragma unroll
  for (int r = 0; r < 8; ++r) { rmax[r] = NEG_INF; rsum[r] = 0.f; s1[r] = 0.f; s2[r] = 0.f; }
  const _Float16* catb = catH + (size_t)b * 512 * NPTS;
  for (int nt = 0; nt < NPTS / 16; ++nt) {
    __syncthreads();
    for (int e = tid; e < 512 * 16; e += 128) {
      int c = e & 511;       // K index
      int jj = e >> 9;       // column
      int kc = c >> 5, kin = c & 31;
      Bp[(kc * 16 + jj) * 32 + kin] = catb[(size_t)c * NPTS + nt * 16 + jj];
    }
    __syncthreads();
    if (nt + 1 < NPTS / 16)
      __builtin_prefetch(catb + (nt + 1) * 16, 0, 2);
    v8f acc = {};
#pragma unroll
    for (int kc = 0; kc < 16; ++kc) {
      v16h bf = *(const v16h*)(Bp + (size_t)(kc * 16 + n16) * 32 + khalf * 16);
      acc = wmma_f16(afr[kc], bf, acc);
    }
#pragma unroll
    for (int r = 0; r < 8; ++r) {
      float v = acc[r];
      if (pass == 0) { s1[r] += v; s2[r] += v * v; }
      else {
        int o = row0 + r + (khalf << 3);
        float y = lrelu((v - mean[o]) * scale[o]);
        rmax[r] = fmaxf(rmax[r], y);
        rsum[r] += y;
      }
    }
  }
  if (pass == 0) {
#pragma unroll
    for (int r = 0; r < 8; ++r) {
      int o = row0 + r + (khalf << 3);
      atomicAdd(&gsum[o], s1[r]);
      atomicAdd(&gsumsq[o], s2[r]);
    }
  } else {
#pragma unroll
    for (int r = 0; r < 8; ++r) {
      float mx = rmax[r], sm = rsum[r];
      for (int off = 8; off >= 1; off >>= 1) {
        mx = fmaxf(mx, __shfl_xor(mx, off, 16));
        sm += __shfl_xor(sm, off, 16);
      }
      if (n16 == 0) {
        int o = row0 + r + (khalf << 3);
        g[b * 2048 + o] = mx;
        g[b * 2048 + 1024 + o] = sm * (1.f / 1024.f);
      }
    }
  }
}

// ---- small classifier pieces ------------------------------------------------
__global__ void k_dense(const float* __restrict__ in, int indim,
                        const float* __restrict__ W, const float* __restrict__ bias,
                        float* __restrict__ out, int odim) {
  int t = blockIdx.x * blockDim.x + threadIdx.x;
  if (t >= 16 * odim) return;
  int o = t % odim, r = t / odim;
  const float* ip = in + (size_t)r * indim;
  const float* wp = W + (size_t)o * indim;
  float acc = bias ? bias[o] : 0.f;
  for (int i = 0; i < indim; ++i) acc += ip[i] * wp[i];
  out[t] = acc;
}

__global__ void k_bnrows(float* buf, int cols) {
  int o = blockIdx.x * blockDim.x + threadIdx.x;
  if (o >= cols) return;
  float s = 0.f, s2 = 0.f;
  for (int r = 0; r < 16; ++r) { float v = buf[r * cols + o]; s += v; s2 += v * v; }
  float m = s * (1.f / 16.f);
  float var = s2 * (1.f / 16.f) - m * m;
  float sc = rsqrtf(var + EPSV);
  for (int r = 0; r < 16; ++r) buf[r * cols + o] = lrelu((buf[r * cols + o] - m) * sc);
}

__global__ void k_heads(const float* __restrict__ mlog, const float* __restrict__ blog,
                        float* __restrict__ out) {
  int r = threadIdx.x;
  if (r >= 16) return;
  float mx = NEG_INF;
  for (int o = 0; o < 40; ++o) mx = fmaxf(mx, mlog[r * 40 + o]);
  float s = 0.f;
  for (int o = 0; o < 40; ++o) s += expf(mlog[r * 40 + o] - mx);
  float lse = mx + logf(s);
  for (int o = 0; o < 40; ++o) out[r * 40 + o] = mlog[r * 40 + o] - lse;
  float tmp[40];
  for (int o = 0; o < 40; ++o) {
    float m2 = NEG_INF;
    for (int jj = 0; jj < 5; ++jj) m2 = fmaxf(m2, blog[jj * 640 + r * 40 + o]);
    float s2 = 0.f;
    for (int jj = 0; jj < 5; ++jj) s2 += expf(blog[jj * 640 + r * 40 + o] - m2);
    tmp[o] = m2 + logf(s2);
  }
  float mx2 = NEG_INF;
  for (int o = 0; o < 40; ++o) mx2 = fmaxf(mx2, tmp[o]);
  float s3 = 0.f;
  for (int o = 0; o < 40; ++o) s3 += expf(tmp[o] - mx2);
  float lse2 = mx2 + logf(s3);
  for (int o = 0; o < 40; ++o) out[640 + r * 40 + o] = tmp[o] - lse2;
}

// (J,A,Kd) -> (A,J,Kd)
__global__ void k_trJ(const float* __restrict__ in, int J, int A, int Kd,
                      float* __restrict__ out) {
  long t = (long)blockIdx.x * blockDim.x + threadIdx.x;
  long total = (long)J * A * Kd;
  if (t >= total) return;
  int kk = (int)(t % Kd);
  int a = (int)((t / Kd) % A);
  int jj = (int)(t / ((long)Kd * A));
  out[((size_t)a * J + jj) * Kd + kk] = in[t];
}

// ---- host -------------------------------------------------------------------
extern "C" void kernel_launch(void* const* d_in, const int* in_sizes, int n_in,
                              void* d_out, int out_size, void* d_ws, size_t ws_size,
                              hipStream_t stream) {
  (void)in_sizes; (void)n_in; (void)out_size; (void)ws_size;
  const float* x   = (const float*)d_in[0];
  const float* W1  = (const float*)d_in[1];
  const float* W2  = (const float*)d_in[2];
  const float* W3  = (const float*)d_in[3];
  const float* W4  = (const float*)d_in[4];
  const float* W5  = (const float*)d_in[5];
  const float* Wg0 = (const float*)d_in[6];
  const float* Wg4 = (const float*)d_in[7];
  const float* bg4 = (const float*)d_in[8];
  const float* Wg8 = (const float*)d_in[9];
  const float* bg8 = (const float*)d_in[10];
  const float* Wb0 = (const float*)d_in[11];
  const float* Wb4 = (const float*)d_in[12];
  const float* bb4 = (const float*)d_in[13];
  const float* Wb8 = (const float*)d_in[14];
  const float* bb8 = (const float*)d_in[15];
  float* out = (float*)d_out;

  float* ws = (float*)d_ws;
  size_t off = 0;
  float* xtA    = ws + off; off += (size_t)16 * 1024 * 256;
  float* xtB    = ws + off; off += (size_t)16 * 1024 * 256;
  float* hmax   = ws + off; off += (size_t)16 * 256 * 1024;
  _Float16* catH = (_Float16*)(ws + off); off += (size_t)16 * 512 * 1024 / 2;
  float* gsum   = ws + off; off += 1024;
  float* gsq    = ws + off; off += 1024;
  float* bmean  = ws + off; off += 1024;
  float* bscale = ws + off; off += 1024;
  float* g      = ws + off; off += 16 * 2048;
  float* h1     = ws + off; off += 16 * 512;
  float* h2     = ws + off; off += 16 * 256;
  float* mlog   = ws + off; off += 16 * 40;
  float* blog   = ws + off; off += 5 * 16 * 40;
  // packed f16 weights (fragment-ready)
  _Float16* Wp1 = (_Float16*)(ws + off); off += 4 * 1 * 512 / 2;
  _Float16* Wp2 = (_Float16*)(ws + off); off += 4 * 4 * 512 / 2;
  _Float16* Wp3 = (_Float16*)(ws + off); off += 8 * 4 * 512 / 2;
  _Float16* Wp4 = (_Float16*)(ws + off); off += 16 * 8 * 512 / 2;
  _Float16* Wp5 = (_Float16*)(ws + off); off += 64 * 16 * 512 / 2;
  int*   idx    = (int*)(ws + off);

  // pack weights into WMMA A-fragment order
  k_packA<<<(4 * 1 * 512 + 255) / 256, 256, 0, stream>>>(W1, 10, 1, Wp1, 4 * 1 * 512);
  k_packA<<<(4 * 4 * 512 + 255) / 256, 256, 0, stream>>>(W2, 128, 4, Wp2, 4 * 4 * 512);
  k_packA<<<(8 * 4 * 512 + 255) / 256, 256, 0, stream>>>(W3, 128, 4, Wp3, 8 * 4 * 512);
  k_packA<<<(16 * 8 * 512 + 255) / 256, 256, 0, stream>>>(W4, 256, 8, Wp4, 16 * 8 * 512);
  k_packA<<<(64 * 16 * 512 + 255) / 256, 256, 0, stream>>>(W5, 512, 16, Wp5, 64 * 16 * 512);

  // x -> xt0 (B,N,5)
  k_xt0<<<(16 * 5 * 1024 + 255) / 256, 256, 0, stream>>>(x, xtA);

  const float ecCount = (float)(16 * 1024 * 20);
  dim3 gGemm(320, 16);

  // ---- layer 1: C=5 -> O=64, cat offset 0
  k_knn<<<256, 64, 0, stream>>>(xtA, 5, idx);
  k_fill<<<(16 * 64 * 1024 + 255) / 256, 256, 0, stream>>>(hmax, (long)16 * 64 * 1024, NEG_INF);
  k_fill<<<8, 256, 0, stream>>>(gsum, 2048, 0.f);
  k_ecgemm<1><<<gGemm, 128, 0, stream>>>(xtA, 5, Wp1, 64, idx, hmax, gsum, gsq);
  k_bnstats<<<1, 64, 0, stream>>>(gsum, gsq, ecCount, 64, bmean, bscale);
  k_ecapply<<<(16 * 64 * 1024 + 255) / 256, 256, 0, stream>>>(hmax, 64, bmean, bscale, catH, 0, xtB, 1);

  // ---- layer 2: C=64 -> O=64, cat offset 64
  k_knn<<<256, 64, 0, stream>>>(xtB, 64, idx);
  k_fill<<<(16 * 64 * 1024 + 255) / 256, 256, 0, stream>>>(hmax, (long)16 * 64 * 1024, NEG_INF);
  k_fill<<<8, 256, 0, stream>>>(gsum, 2048, 0.f);
  k_ecgemm<4><<<gGemm, 128, 0, stream>>>(xtB, 64, Wp2, 64, idx, hmax, gsum, gsq);
  k_bnstats<<<1, 64, 0, stream>>>(gsum, gsq, ecCount, 64, bmean, bscale);
  k_ecapply<<<(16 * 64 * 1024 + 255) / 256, 256, 0, stream>>>(hmax, 64, bmean, bscale, catH, 64, xtA, 1);

  // ---- layer 3: C=64 -> O=128, cat offset 128
  k_knn<<<256, 64, 0, stream>>>(xtA, 64, idx);
  k_fill<<<(16 * 128 * 1024 + 255) / 256, 256, 0, stream>>>(hmax, (long)16 * 128 * 1024, NEG_INF);
  k_fill<<<8, 256, 0, stream>>>(gsum, 2048, 0.f);
  k_ecgemm<4><<<gGemm, 128, 0, stream>>>(xtA, 64, Wp3, 128, idx, hmax, gsum, gsq);
  k_bnstats<<<2, 64, 0, stream>>>(gsum, gsq, ecCount, 128, bmean, bscale);
  k_ecapply<<<(16 * 128 * 1024 + 255) / 256, 256, 0, stream>>>(hmax, 128, bmean, bscale, catH, 128, xtB, 1);

  // ---- layer 4: C=128 -> O=256, cat offset 256
  k_knn<<<256, 64, 0, stream>>>(xtB, 128, idx);
  k_fill<<<(16 * 256 * 1024 + 255) / 256, 256, 0, stream>>>(hmax, (long)16 * 256 * 1024, NEG_INF);
  k_fill<<<8, 256, 0, stream>>>(gsum, 2048, 0.f);
  k_ecgemm<8><<<gGemm, 128, 0, stream>>>(xtB, 128, Wp4, 256, idx, hmax, gsum, gsq);
  k_bnstats<<<4, 64, 0, stream>>>(gsum, gsq, ecCount, 256, bmean, bscale);
  k_ecapply<<<(16 * 256 * 1024 + 255) / 256, 256, 0, stream>>>(hmax, 256, bmean, bscale, catH, 256, xtA, 0);

  // ---- W5 (1024x512) pointwise conv, BN over (b,n), max+mean pooling -> g
  k_fill<<<8, 256, 0, stream>>>(gsum, 2048, 0.f);
  k_w5<<<dim3(16, 16), 128, 0, stream>>>(catH, Wp5, 0, bmean, bscale, gsum, gsq, g);
  k_bnstats<<<16, 64, 0, stream>>>(gsum, gsq, (float)(16 * 1024), 1024, bmean, bscale);
  k_w5<<<dim3(16, 16), 128, 0, stream>>>(catH, Wp5, 1, bmean, bscale, gsum, gsq, g);

  // ---- main classifier head
  k_dense<<<(16 * 512 + 127) / 128, 128, 0, stream>>>(g, 2048, Wg0, nullptr, h1, 512);
  k_bnrows<<<8, 64, 0, stream>>>(h1, 512);
  k_dense<<<(16 * 256 + 127) / 128, 128, 0, stream>>>(h1, 512, Wg4, bg4, h2, 256);
  k_bnrows<<<4, 64, 0, stream>>>(h2, 256);
  k_dense<<<(16 * 40 + 127) / 128, 128, 0, stream>>>(h2, 256, Wg8, bg8, mlog, 40);

  // ---- 5 branch classifier heads
  for (int j = 0; j < 5; ++j) {
    k_dense<<<(16 * 512 + 127) / 128, 128, 0, stream>>>(
        g, 2048, Wb0 + (size_t)j * 512 * 2048, nullptr, h1, 512);
    k_bnrows<<<8, 64, 0, stream>>>(h1, 512);
    k_dense<<<(16 * 256 + 127) / 128, 128, 0, stream>>>(
        h1, 512, Wb4 + (size_t)j * 256 * 512, bb4 + (size_t)j * 256, h2, 256);
    k_bnrows<<<4, 64, 0, stream>>>(h2, 256);
    k_dense<<<(16 * 40 + 127) / 128, 128, 0, stream>>>(
        h2, 256, Wb8 + (size_t)j * 40 * 256, bb8 + (size_t)j * 40, blog + j * 640, 40);
  }

  // ---- heads: log_softmax(main), log_softmax(logsumexp(branches))
  k_heads<<<1, 32, 0, stream>>>(mlog, blog, out);

  // ---- pass-through outputs
  hipMemcpyAsync(out + 1280,    Wg0, (size_t)512 * 2048 * 4, hipMemcpyDeviceToDevice, stream);
  hipMemcpyAsync(out + 1049856, Wg4, (size_t)256 * 512 * 4,  hipMemcpyDeviceToDevice, stream);
  hipMemcpyAsync(out + 1180928, Wg8, (size_t)40 * 256 * 4,   hipMemcpyDeviceToDevice, stream);
  k_trJ<<<(int)(((long)5 * 512 * 2048 + 255) / 256), 256, 0, stream>>>(Wb0, 5, 512, 2048, out + 1191168);
  k_trJ<<<(int)(((long)5 * 256 * 512 + 255) / 256), 256, 0, stream>>>(Wb4, 5, 256, 512, out + 6434048);
  k_trJ<<<(int)(((long)5 * 40 * 256 + 255) / 256), 256, 0, stream>>>(Wb8, 5, 40, 256, out + 7089408);
}
